// ScaledDotProductAttention_30691836297267
// MI455X (gfx1250) — matrix operand order repaired and checked by reading, not compiled, not run on hardware
//
#include <hip/hip_runtime.h>

typedef _Float16 v2h  __attribute__((ext_vector_type(2)));
typedef _Float16 v8h  __attribute__((ext_vector_type(8)));
typedef _Float16 v16h __attribute__((ext_vector_type(16)));
typedef float    v8f  __attribute__((ext_vector_type(8)));
typedef float    v4f  __attribute__((ext_vector_type(4)));

#define WAVES  8
#define QTILE  16
#define KVTILE 64
#define DHEAD  128
#define SEQ    2048

// Build a 16-half WMMA operand from two contiguous 16-byte LDS chunks.
__device__ __forceinline__ v16h ld16h(const _Float16* p0, const _Float16* p1) {
    v8h lo = *(const v8h*)p0;
    v8h hi = *(const v8h*)p1;
    return __builtin_shufflevector(lo, hi, 0,1,2,3,4,5,6,7,8,9,10,11,12,13,14,15);
}

__global__ __launch_bounds__(256)
void fa_causal_f16wmma(const float* __restrict__ Q, const float* __restrict__ K,
                       const float* __restrict__ V, float* __restrict__ O) {
    // QPs: Q staging tile early (16x128 per wave), P scratch (16x64 per wave) after
    // the loop's first barrier -- aliasing is barrier-ordered and safe.
    __shared__ __attribute__((aligned(32))) _Float16 QPs[WAVES][QTILE * DHEAD];   // 32 KB
    __shared__ __attribute__((aligned(32))) _Float16 Ks[KVTILE * DHEAD];          // 16 KB
    __shared__ __attribute__((aligned(32))) _Float16 VTs[DHEAD * KVTILE];         // 16 KB (V transposed)

    const int tid  = threadIdx.x;
    const int wave = tid >> 5;
    const int lane = tid & 31;
    const int nidx = lane & 15;
    const int hi   = lane >> 4;              // 0 or 1 (lane half)
    const int b    = blockIdx.y;
    const int qbase_blk = blockIdx.x * 128;
    const int qbase = qbase_blk + wave * QTILE;

    // 1/sqrt(128) * log2(e): softmax runs in exp2 domain (v_exp_f32 is base-2 native)
    const float scale = 0.12751744782021648f;

    // ---- stage this wave's 16x128 Q tile into LDS as scaled f16 ----
    {
        const int row  = lane & 15;
        const int colb = (lane >> 4) * 64;   // 0 or 64
        const float* gq = Q + ((size_t)(b * SEQ + qbase + row)) * DHEAD + colb;
        _Float16* sq = &QPs[wave][row * DHEAD + colb];
        #pragma unroll
        for (int c = 0; c < 4; ++c) {
            v4f f0 = *(const v4f*)(gq + c * 16 + 0);
            v4f f1 = *(const v4f*)(gq + c * 16 + 4);
            v4f f2 = *(const v4f*)(gq + c * 16 + 8);
            v4f f3 = *(const v4f*)(gq + c * 16 + 12);
            v8h h0, h1;
            #pragma unroll
            for (int e = 0; e < 4; ++e) {
                h0[e]     = (_Float16)(f0[e] * scale);
                h0[e + 4] = (_Float16)(f1[e] * scale);
                h1[e]     = (_Float16)(f2[e] * scale);
                h1[e + 4] = (_Float16)(f3[e] * scale);
            }
            *(v8h*)(sq + c * 16 + 0) = h0;
            *(v8h*)(sq + c * 16 + 8) = h1;
        }
    }
    __syncthreads();

    // ---- Q as WMMA A operands (16x32 f16 layout), kept in VGPRs ----
    v16h a_q[4];
    {
        const int m = nidx;
        #pragma unroll
        for (int dc = 0; dc < 4; ++dc) {
            const _Float16* p = &QPs[wave][m * DHEAD + dc * 32 + hi * 8];
            a_q[dc] = ld16h(p, p + 16);
        }
    }

    // all-ones B operand: row-sum of P via WMMA (replicated across lanes in C-layout)
    v16h ones16;
    #pragma unroll
    for (int e = 0; e < 16; ++e) ones16[e] = (_Float16)1.0f;

    v8f  c_o[8];
    v8f  c_l;                                // running softmax denominator per row
    float m_run[8];
    #pragma unroll
    for (int j = 0; j < 8; ++j)
        #pragma unroll
        for (int i = 0; i < 8; ++i) c_o[j][i] = 0.0f;
    #pragma unroll
    for (int i = 0; i < 8; ++i) { c_l[i] = 0.0f; m_run[i] = -1e30f; }

    const int q_hi   = qbase + QTILE - 1;
    const int ntiles = (qbase_blk + 128) / KVTILE;   // block-uniform causal bound

    // staging mapping (shared by load + prefetch)
    const int r0   = (tid & 31) * 2;       // kv rows r0, r0+1
    const int colb = (tid >> 5) * 16;      // 16-wide d slice

    for (int kt = 0; kt < ntiles; ++kt) {
        const int kbase = kt * KVTILE;
        __syncthreads();   // previous iteration's LDS reads complete
        // ---- cooperative stage: K row-major f16, V transposed f16 (kv-pairs packed) ----
        {
            const float* gk = K + ((size_t)(b * SEQ + kbase + r0)) * DHEAD + colb;
            const float* gv = V + ((size_t)(b * SEQ + kbase + r0)) * DHEAD + colb;
            #pragma unroll
            for (int r = 0; r < 2; ++r) {
                _Float16* sk = &Ks[(r0 + r) * DHEAD + colb];
                #pragma unroll
                for (int c = 0; c < 2; ++c) {
                    v4f f0 = *(const v4f*)(gk + r * DHEAD + c * 8 + 0);
                    v4f f1 = *(const v4f*)(gk + r * DHEAD + c * 8 + 4);
                    v8h h;
                    #pragma unroll
                    for (int e = 0; e < 4; ++e) { h[e] = (_Float16)f0[e]; h[e + 4] = (_Float16)f1[e]; }
                    *(v8h*)(sk + c * 8) = h;
                }
            }
            v4f a0[4], a1[4];
            #pragma unroll
            for (int c = 0; c < 4; ++c) {
                a0[c] = *(const v4f*)(gv + c * 4);
                a1[c] = *(const v4f*)(gv + DHEAD + c * 4);
            }
            #pragma unroll
            for (int c = 0; c < 4; ++c)
                #pragma unroll
                for (int e = 0; e < 4; ++e) {
                    v2h pr; pr[0] = (_Float16)a0[c][e]; pr[1] = (_Float16)a1[c][e];
                    *(v2h*)&VTs[(colb + c * 4 + e) * KVTILE + r0] = pr;
                }
        }
        __syncthreads();

        // ---- prefetch next KV tile into cache while this tile computes ----
        if (kt + 1 < ntiles) {
            const int kb2 = kbase + KVTILE;
            const float* pk = K + ((size_t)(b * SEQ + kb2 + r0)) * DHEAD + colb;
            const float* pv = V + ((size_t)(b * SEQ + kb2 + r0)) * DHEAD + colb;
            __builtin_prefetch(pk, 0, 3);
            __builtin_prefetch(pk + DHEAD, 0, 3);
            __builtin_prefetch(pv, 0, 3);
            __builtin_prefetch(pv + DHEAD, 0, 3);
        }

        if (kbase > q_hi) continue;            // wave-uniform skip above diagonal

        // ---- S = Q K^T : four 16x16 f32 tiles via 16 WMMAs ----
        v8f c_s[4];
        #pragma unroll
        for (int t = 0; t < 4; ++t) {
            v8f acc;
            #pragma unroll
            for (int i = 0; i < 8; ++i) acc[i] = 0.0f;
            #pragma unroll
            for (int dc = 0; dc < 4; ++dc) {
                const _Float16* p = &Ks[(t * 16 + nidx) * DHEAD + dc * 32 + hi * 8];
                v16h bk = ld16h(p, p + 16);
                acc = __builtin_amdgcn_wmma_f32_16x16x32_f16(false, a_q[dc], false, bk,
                                                             (short)0, acc, false, false);
            }
            c_s[t] = acc;
        }

        // ---- causal mask (per 16-key sub-tile, scalar-gated) ----
        #pragma unroll
        for (int t = 0; t < 4; ++t) {
            if (kbase + t * 16 + 15 > qbase) {
                const int kv = kbase + t * 16 + nidx;
                #pragma unroll
                for (int i = 0; i < 8; ++i) {
                    const int qi = qbase + i + hi * 8;
                    if (kv > qi) c_s[t][i] = -1e30f;
                }
            }
        }

        // ---- online softmax (exp2 domain, raw v_exp_f32); P overwrites c_s ----
        #pragma unroll
        for (int i = 0; i < 8; ++i) {
            float v = fmaxf(fmaxf(c_s[0][i], c_s[1][i]), fmaxf(c_s[2][i], c_s[3][i]));
            v = fmaxf(v, __shfl_xor(v, 8, 32));
            v = fmaxf(v, __shfl_xor(v, 4, 32));
            v = fmaxf(v, __shfl_xor(v, 2, 32));
            v = fmaxf(v, __shfl_xor(v, 1, 32));
            const float m_new = fmaxf(m_run[i], v);
            const float alpha = __builtin_amdgcn_exp2f(m_run[i] - m_new);
            m_run[i] = m_new;
            #pragma unroll
            for (int t = 0; t < 4; ++t) c_s[t][i] = __builtin_amdgcn_exp2f(c_s[t][i] - m_new);
            c_l[i] *= alpha;
            #pragma unroll
            for (int j = 0; j < 8; ++j) c_o[j][i] *= alpha;
        }

        // ---- P: C-layout -> A-layout through per-wave LDS scratch (aliases Q buffer) ----
        {
            _Float16* pp = &QPs[wave][0];
            #pragma unroll
            for (int t = 0; t < 4; ++t)
                #pragma unroll
                for (int i = 0; i < 8; ++i)
                    pp[(i + hi * 8) * KVTILE + t * 16 + nidx] = (_Float16)c_s[t][i];
        }
        asm volatile("s_wait_dscnt 0" ::: "memory");   // wave-local store->load ordering
        v16h a_p[2];
        #pragma unroll
        for (int ch = 0; ch < 2; ++ch) {
            const _Float16* p = &QPs[wave][nidx * KVTILE + ch * 32 + hi * 8];
            a_p[ch] = ld16h(p, p + 16);
        }

        // ---- l += P * 1 (row sums via WMMA, replicated across lanes) ----
        #pragma unroll
        for (int ch = 0; ch < 2; ++ch)
            c_l = __builtin_amdgcn_wmma_f32_16x16x32_f16(false, a_p[ch], false, ones16,
                                                         (short)0, c_l, false, false);

        // ---- O += P V : 16 WMMAs across the 128-wide head dim ----
        #pragma unroll
        for (int j = 0; j < 8; ++j)
            #pragma unroll
            for (int ch = 0; ch < 2; ++ch) {
                const _Float16* p = &VTs[(j * 16 + nidx) * KVTILE + ch * 32 + hi * 8];
                v16h bv = ld16h(p, p + 16);
                c_o[j] = __builtin_amdgcn_wmma_f32_16x16x32_f16(false, a_p[ch], false, bv,
                                                                (short)0, c_o[j], false, false);
            }
    }

    // ---- normalize and store (coalesced b32 stores across each 16-lane half) ----
    #pragma unroll
    for (int i = 0; i < 8; ++i) {
        const float inv = 1.0f / c_l[i];
        const int row = qbase + i + hi * 8;
        float* go = O + ((size_t)(b * SEQ + row)) * DHEAD + nidx;
        #pragma unroll
        for (int j = 0; j < 8; ++j)
            go[j * 16] = c_o[j][i] * inv;
    }
}

extern "C" void kernel_launch(void* const* d_in, const int* in_sizes, int n_in,
                              void* d_out, int out_size, void* d_ws, size_t ws_size,
                              hipStream_t stream) {
    (void)in_sizes; (void)n_in; (void)out_size; (void)d_ws; (void)ws_size;
    const float* q = (const float*)d_in[0];
    const float* k = (const float*)d_in[1];
    const float* v = (const float*)d_in[2];
    float* o = (float*)d_out;
    dim3 grid(SEQ / 128, 16);   // (query blocks, batch)
    dim3 block(256);
    fa_causal_f16wmma<<<grid, block, 0, stream>>>(q, k, v, o);
}
